// RelEncoding_40303973106096
// MI455X (gfx1250) — compile-verified, matrix operand order checked by reading, not run on hardware
//
#include <hip/hip_runtime.h>

typedef __attribute__((ext_vector_type(16))) _Float16 v16h;
typedef __attribute__((ext_vector_type(8)))  float    v8f;
typedef __attribute__((ext_vector_type(4)))  float    f4;

#define BH_   32
#define L_    2048
#define D_    64
#define ROWS_ 257           // 2*MAX_REL_POS + 1
#define NT_   17            // ceil(257/16) N-tiles
#define NPAD_ (NT_ * 16)    // 272

// Per-lane fragment load for 16-bit WMMA A/B operands (CDNA5 ISA):
// lane = M-row (A) / N-col (B); half-slots h=0..7 hold K = 8*kh .. 8*kh+7,
// h=8..15 hold K = 16+8*kh .. 23+8*kh  (kh = lane>>4). Two contiguous 32-byte
// chunks -> four global_load_b128, then packed f32->f16 converts.
__device__ __forceinline__ v16h load_frag_f16(const float* __restrict__ row,
                                              int kh, int kbase) {
    const f4* p4 = (const f4*)(row + kbase);
    f4 q0 = p4[2 * kh + 0];
    f4 q1 = p4[2 * kh + 1];
    f4 q2 = p4[2 * kh + 4];
    f4 q3 = p4[2 * kh + 5];
    v16h f;
    f[0]  = (_Float16)q0.x; f[1]  = (_Float16)q0.y;
    f[2]  = (_Float16)q0.z; f[3]  = (_Float16)q0.w;
    f[4]  = (_Float16)q1.x; f[5]  = (_Float16)q1.y;
    f[6]  = (_Float16)q1.z; f[7]  = (_Float16)q1.w;
    f[8]  = (_Float16)q2.x; f[9]  = (_Float16)q2.y;
    f[10] = (_Float16)q2.z; f[11] = (_Float16)q2.w;
    f[12] = (_Float16)q3.x; f[13] = (_Float16)q3.y;
    f[14] = (_Float16)q3.z; f[15] = (_Float16)q3.w;
    return f;
}

// One 16x16 output tile of S = A(16xD) * W^T(D x N): two chained K=32 WMMAs.
// Weight row index clamped to 256: padded S columns 257..271 hold garbage but
// are never read by the expansion phase (clamped index <= 256), so no zeroing
// and no per-lane predication is needed.
__device__ __forceinline__ void compute_tile(int nt,
                                             const float* __restrict__ weight,
                                             v16h a0, v16h a1,
                                             float (*S)[NPAD_],
                                             int sub, int kh) {
    int r = nt * 16 + sub;
    r = (r < ROWS_) ? r : (ROWS_ - 1);
    const float* wrow = weight + (size_t)r * D_;
    v16h b0 = load_frag_f16(wrow, kh, 0);
    v16h b1 = load_frag_f16(wrow, kh, 32);
    v8f c = {};
    c = __builtin_amdgcn_wmma_f32_16x16x32_f16(false, a0, false, b0,
                                               (short)0, c, false, false);
    c = __builtin_amdgcn_wmma_f32_16x16x32_f16(false, a1, false, b1,
                                               (short)0, c, false, false);
    // C layout: VGPR j -> M = j + 8*(lane>=16), N = lane&15
#pragma unroll
    for (int j = 0; j < 8; ++j)
        S[j + 8 * kh][nt * 16 + sub] = c[j];
}

__global__ __launch_bounds__(256)
void rel_enc_fused_kernel(const float* __restrict__ inpt,
                          const float* __restrict__ weight,
                          float* __restrict__ out)
{
    __shared__ float S[16][NPAD_];   // S[i][r] = inpt[b,l0+i,:] . weight[r,:]

    const int b    = blockIdx.x;      // 0..31
    const int l0   = blockIdx.y * 16; // 0..2047 step 16
    const int tid  = threadIdx.x;
    const int lane = tid & 31;        // wave32
    const int wave = tid >> 5;        // 0..7
    const int sub  = lane & 15;       // M-row (A) / N-col (B) within fragment
    const int kh   = lane >> 4;       // which K-half this lane holds

    // ---- A fragments: row l0+sub of inpt, fp32 -> f16, K = 0..31 / 32..63
    const float* arow = inpt + ((size_t)b * L_ + (size_t)(l0 + sub)) * D_;
    v16h a0 = load_frag_f16(arow, kh, 0);
    v16h a1 = load_frag_f16(arow, kh, 32);

    // ---- GEMM phase: 17 N-tiles; waves 0..7 take {w, w+8}, wave 0 takes 16.
    compute_tile(wave,     weight, a0, a1, S, sub, kh);
    compute_tile(wave + 8, weight, a0, a1, S, sub, kh);
    if (wave == 0)                          // wave-uniform: EXEC stays all-1s
        compute_tile(16, weight, a0, a1, S, sub, kh);
    __syncthreads();

    // ---- Expansion: out[b, l0+i, m] = S[i][clamp(m-(l0+i),-128,128)+128] ----
    // 16 rows x 2048 cols = 8192 float4 stores per block, coalesced,
    // non-temporal (output is write-once, 2.7x the size of L2).
    float* outbase = out + ((size_t)b * L_ + (size_t)l0) * L_;
    for (int e = tid; e < 16 * (L_ / 4); e += 256) {
        const int i  = e >> 9;              // L_/4 == 512
        const int m0 = (e & 511) << 2;
        const int l  = l0 + i;
        const float* srow = S[i];
        const int t = m0 - l;               // relative distance of element 0
        int i0 = t + 0; i0 = i0 < -128 ? 0 : (i0 > 128 ? 256 : i0 + 128);
        int i1 = t + 1; i1 = i1 < -128 ? 0 : (i1 > 128 ? 256 : i1 + 128);
        int i2 = t + 2; i2 = i2 < -128 ? 0 : (i2 > 128 ? 256 : i2 + 128);
        int i3 = t + 3; i3 = i3 < -128 ? 0 : (i3 > 128 ? 256 : i3 + 128);
        f4 v;
        v.x = srow[i0];
        v.y = srow[i1];
        v.z = srow[i2];
        v.w = srow[i3];
        __builtin_nontemporal_store(v, (f4*)(outbase + (size_t)i * L_ + m0));
    }
}

extern "C" void kernel_launch(void* const* d_in, const int* in_sizes, int n_in,
                              void* d_out, int out_size, void* d_ws, size_t ws_size,
                              hipStream_t stream)
{
    const float* inpt   = (const float*)d_in[0];  // [32, 2048, 64]
    const float* weight = (const float*)d_in[1];  // [257, 64]
    // d_in[2] = max_seq_len (2048), fixed at compile time
    float* out = (float*)d_out;                   // [32, 2048, 2048]

    dim3 grid(BH_, L_ / 16);
    dim3 block(256);
    rel_enc_fused_kernel<<<grid, block, 0, stream>>>(inpt, weight, out);
}
